// CustomLSTM_78486232367302
// MI455X (gfx1250) — matrix operand order repaired and checked by reading, not compiled, not run on hardware
//
#include <hip/hip_runtime.h>
#include <hip/hip_bf16.h>
#include <hip/hip_fp16.h>

// ---------------- problem constants ----------------
#define BDIM 256
#define TDIM 1024
#define IDIM 64
#define HDIM 512
#define DDIM (IDIM + 2 * HDIM)   // 1088
#define MT 16                    // batch rows per workgroup
#define NWAVES 16
#define NTHREADS (NWAVES * 32)
#define HCs 1096                 // hc row stride in halves (padded: conflict-free)
#define HS 516                   // f32 state row stride in floats (padded)

typedef __attribute__((ext_vector_type(16))) _Float16 v16h;
typedef __attribute__((ext_vector_type(8)))  _Float16 h8;
typedef __attribute__((ext_vector_type(8)))  float    v8f;

union Frag16 { v16h v; h8 h[2]; };

// ---------------- CDNA5 feature probes ----------------
#if defined(__has_builtin)
#if __has_builtin(__builtin_amdgcn_global_load_async_to_lds_b64)
#define HAVE_ASYNC_LD 1
#endif
#endif

#if defined(__has_builtin) && __has_builtin(__builtin_amdgcn_s_wait_asynccnt)
#define WAIT_ASYNC() __builtin_amdgcn_s_wait_asynccnt(0)
#else
#define WAIT_ASYNC() asm volatile("s_wait_asynccnt 0" ::: "memory")
#endif

#ifdef HAVE_ASYNC_LD
// Builtin signature (probe-confirmed): (global int2* src, lds ptr dst, imm offset, imm cpol)
typedef int async_i2 __attribute__((vector_size(8)));
typedef __attribute__((address_space(1))) async_i2* async_gptr;
typedef __attribute__((address_space(3))) async_i2* async_lptr;
#endif

// Fast activations: approximate rcp (single v_rcp_f32, no IEEE div fixup chain).
__device__ __forceinline__ float sigm(float x) {
    return __builtin_amdgcn_rcpf(1.f + __expf(-x));
}
__device__ __forceinline__ float tanh_fast(float x) {
#if defined(__has_builtin) && __has_builtin(__builtin_amdgcn_tanhf)
    return __builtin_amdgcn_tanhf(x);          // native V_TANH_F32 on CDNA5
#elif defined(__has_builtin) && __has_builtin(__builtin_amdgcn_tanh_f32)
    return __builtin_amdgcn_tanh_f32(x);
#else
    x = fminf(fmaxf(x, -15.f), 15.f);
    float e = __expf(2.f * x);
    return (e - 1.f) * __builtin_amdgcn_rcpf(e + 1.f);
#endif
}

// ---------------- weight conversion (f32 -> f16, once per launch) ----------------
__global__ void cvt_weights(const float* __restrict__ We, const float* __restrict__ Wi,
                            const float* __restrict__ Wf, const float* __restrict__ Wc,
                            const float* __restrict__ Wo, _Float16* __restrict__ ws) {
    const size_t nWe = (size_t)HDIM * HDIM;
    const size_t nWg = (size_t)HDIM * DDIM;
    const size_t total = nWe + 4 * nWg;
    for (size_t i = (size_t)blockIdx.x * blockDim.x + threadIdx.x; i < total;
         i += (size_t)gridDim.x * blockDim.x) {
        float v;
        if (i < nWe) {
            v = We[i];
        } else {
            size_t j = i - nWe;
            size_t g = j / nWg, r = j % nWg;
            const float* W = (g == 0) ? Wi : (g == 1) ? Wf : (g == 2) ? Wc : Wo;
            v = W[r];
        }
        ws[i] = (_Float16)v;
    }
}

// ---------------- persistent recurrent kernel ----------------
__launch_bounds__(NTHREADS, 1)
__global__ void lstm_scan(const float* __restrict__ x,
                          const _Float16* __restrict__ Weh,   // [H][H] f16
                          const _Float16* __restrict__ Wgh,   // [4][H][D] f16 (i,f,c,o)
                          const float* __restrict__ b_i, const float* __restrict__ b_f,
                          const float* __restrict__ b_c, const float* __restrict__ b_o,
                          const float* __restrict__ ln_g, const float* __restrict__ ln_b,
                          const float* __restrict__ W_fc,
                          float* __restrict__ out) {
    __shared__ _Float16 hc[MT * HCs];      // [xt | tanh(h_en@We^T) | layernorm(h)] as f16
    __shared__ float hS[MT * HS];          // h (recurrent); holds pre-norm u between C and D
    __shared__ float cS[MT * HS];          // c
    __shared__ float heS[MT * HS];         // h_en
    __shared__ float xstage[MT * IDIM];    // staged x(:, t, :)
    __shared__ float pb[4 * HDIM];         // biases i,f,c,o
    __shared__ float plg[HDIM], plb[HDIM];
    __shared__ float pfc[HDIM + 4];
    __shared__ float prev_out[MT], en_s[MT], prev_x[MT], ddS[MT], curxS[MT];

    const int tid  = threadIdx.x;
    const int lane = tid & 31;
    const int wave = tid >> 5;
    const int b0   = blockIdx.x * MT;

    // ---- init state ----
    for (int i = tid; i < MT * HS; i += NTHREADS) { hS[i] = 0.f; cS[i] = 0.f; heS[i] = 0.f; }
    for (int i = tid; i < HDIM; i += NTHREADS) {
        pb[i] = b_i[i]; pb[HDIM + i] = b_f[i]; pb[2 * HDIM + i] = b_c[i]; pb[3 * HDIM + i] = b_o[i];
        plg[i] = ln_g[i]; plb[i] = ln_b[i]; pfc[i] = W_fc[i];
    }
    if (tid == 0) pfc[HDIM] = W_fc[HDIM];
    if (tid < MT) { prev_out[tid] = 0.f; en_s[tid] = 0.f; prev_x[tid] = 0.f; }
    __syncthreads();

    const size_t carry = 2ull * BDIM * TDIM;            // start of carry block in d_out
    const size_t BH    = (size_t)BDIM * HDIM;

    for (int t = 0; t < TDIM; ++t) {
        // ================= Phase A0: stage x(:, t, :) into LDS =================
        {
            const int idx = tid * 2;           // 2 dwords per thread, 1024 total
            const int row = idx >> 6;          // == wave
            const int col = idx & 63;
            const float* gp = x + ((size_t)(b0 + row) * TDIM + t) * IDIM + col;
#ifdef HAVE_ASYNC_LD
            __builtin_amdgcn_global_load_async_to_lds_b64(
                (async_gptr)(gp),
                (async_lptr)&xstage[row * IDIM + col], 0, 0);
            WAIT_ASYNC();
#else
            xstage[row * IDIM + col]     = gp[0];
            xstage[row * IDIM + col + 1] = gp[1];
#endif
            if (t + 1 < TDIM) __builtin_prefetch(gp + IDIM, 0, 1);  // next step's x
        }

        // ================= Phase A1: per-wave row prep (wave == row m) =========
        {
            const int m = wave;
            float s = 0.f, q = 0.f;
            for (int j = 0; j < HDIM / 32; ++j) {
                float v = hS[m * HS + j * 32 + lane];
                s += v; q += v * v;
            }
            for (int off = 16; off; off >>= 1) { s += __shfl_xor(s, off, 32); q += __shfl_xor(q, off, 32); }
            const float mean = s * (1.f / HDIM);
            const float var  = q * (1.f / HDIM) - mean * mean;
            const float rstd = rsqrtf(var + 1e-5f);
            for (int j = 0; j < HDIM / 32; ++j) {
                int n = j * 32 + lane;
                float v = (hS[m * HS + n] - mean) * rstd * plg[n] + plb[n];
                hc[m * HCs + IDIM + HDIM + n] = (_Float16)v;   // h_norm columns
            }
            for (int c2 = lane; c2 < IDIM; c2 += 32)
                hc[m * HCs + c2] = (_Float16)xstage[m * IDIM + c2];  // xt columns
            if (lane == 0) {
                float cx = xstage[m * IDIM];
                curxS[m] = cx;
                ddS[m]   = cx - prev_x[m];
            }
        }

        // ================= Phase B: t_en = tanh(h_en @ We^T) ===================
        {
            const int al = lane & 15, ahi = lane >> 4;
            for (int tt = 0; tt < 2; ++tt) {
                const int nt = wave * 2 + tt;          // 0..31 N-tiles
                v8f acc = {};
                for (int k = 0; k < HDIM; k += 32) {
                    Frag16 a;
                    const float* p0 = &heS[al * HS + k + ahi * 16];
                    h8 lo, hi;
                    #pragma unroll
                    for (int j = 0; j < 8; ++j) { lo[j] = (_Float16)p0[j]; hi[j] = (_Float16)p0[8 + j]; }
                    a.h[0] = lo; a.h[1] = hi;
                    const int nrow = nt * 16 + al;
                    v16h bfr = *(const v16h*)&Weh[(size_t)nrow * HDIM + k + ahi * 16];
                    acc = __builtin_amdgcn_wmma_f32_16x16x32_f16(false, a.v, false, bfr,
                                                                 (short)0, acc, false, false);
                }
                const int nl = lane & 15, chi = lane >> 4;
                #pragma unroll
                for (int r = 0; r < 8; ++r) {
                    int m = r + chi * 8;
                    hc[m * HCs + IDIM + nt * 16 + nl] = (_Float16)tanh_fast(acc[r]);
                }
            }
        }
        __syncthreads();   // hc complete

        // ================= Phase C: 4-gate GEMM + cell update ==================
        {
            const int al = lane & 15, ahi = lane >> 4;
            for (int tt = 0; tt < 2; ++tt) {
                const int nt   = wave * 2 + tt;
                const int nrow = nt * 16 + al;
                v8f a0 = {}, a1 = {}, a2 = {}, a3 = {};
                #pragma unroll 2
                for (int k = 0; k < DDIM; k += 32) {
                    Frag16 a;
                    a.h[0] = *(const h8*)&hc[al * HCs + k + ahi * 16];
                    a.h[1] = *(const h8*)&hc[al * HCs + k + ahi * 16 + 8];
                    const size_t wb = (size_t)nrow * DDIM + k + ahi * 16;
                    v16h bi_ = *(const v16h*)&Wgh[wb];
                    v16h bf_ = *(const v16h*)&Wgh[(size_t)1 * HDIM * DDIM + wb];
                    v16h bc_ = *(const v16h*)&Wgh[(size_t)2 * HDIM * DDIM + wb];
                    v16h bo_ = *(const v16h*)&Wgh[(size_t)3 * HDIM * DDIM + wb];
                    a0 = __builtin_amdgcn_wmma_f32_16x16x32_f16(false, a.v, false, bi_, (short)0, a0, false, false);
                    a1 = __builtin_amdgcn_wmma_f32_16x16x32_f16(false, a.v, false, bf_, (short)0, a1, false, false);
                    a2 = __builtin_amdgcn_wmma_f32_16x16x32_f16(false, a.v, false, bc_, (short)0, a2, false, false);
                    a3 = __builtin_amdgcn_wmma_f32_16x16x32_f16(false, a.v, false, bo_, (short)0, a3, false, false);
                }
                const int nl = lane & 15, chi = lane >> 4;
                const int n  = nt * 16 + nl;
                const float Bi = pb[n], Bf = pb[HDIM + n], Bc = pb[2 * HDIM + n], Bo = pb[3 * HDIM + n];
                #pragma unroll
                for (int r = 0; r < 8; ++r) {
                    const int m = r + chi * 8;
                    const float iv = sigm(a0[r] + Bi);
                    const float fv = sigm(a1[r] + Bf);
                    const float gv = tanh_fast(a2[r] + Bc);
                    const float ov = sigm(a3[r] + Bo);
                    const float cn = fv * cS[m * HS + n] + iv * gv;
                    cS[m * HS + n] = cn;
                    const float hold = hS[m * HS + n];
                    hS[m * HS + n]  = ov * tanh_fast(cn) + hold;     // pre-norm u overwrites h
                    heS[m * HS + n] += hold * 0.5f * ddS[m];         // h_old half of trapezoid
                }
            }
        }
        __syncthreads();   // u, c complete

        // ================= Phase D: layernorm, outputs, trapezoid ==============
        {
            const int m = wave;
            const float ddm = ddS[m];
            float s = 0.f, q = 0.f;
            for (int j = 0; j < HDIM / 32; ++j) {
                float v = hS[m * HS + j * 32 + lane];
                s += v; q += v * v;
            }
            for (int off = 16; off; off >>= 1) { s += __shfl_xor(s, off, 32); q += __shfl_xor(q, off, 32); }
            const float mean = s * (1.f / HDIM);
            const float var  = q * (1.f / HDIM) - mean * mean;
            const float rstd = rsqrtf(var + 1e-5f);
            float po = 0.f;
            for (int j = 0; j < HDIM / 32; ++j) {
                const int n  = j * 32 + lane;
                const float u  = hS[m * HS + n];
                const float hn = (u - mean) * rstd * plg[n] + plb[n];
                hS[m * HS + n] = hn;
                const float he = heS[m * HS + n] + hn * 0.5f * ddm;
                heS[m * HS + n] = he;
                po += hn * pfc[n];
                if (t == TDIM - 1) {
                    const size_t bi2 = (size_t)(b0 + m) * HDIM + n;
                    out[carry + bi2]                 = hn;            // carry h
                    out[carry + BH + bi2]            = cS[m * HS + n];// carry c
                    out[carry + 2 * BH + BDIM + bi2] = he;            // carry h_en
                }
            }
            for (int off = 16; off; off >>= 1) po += __shfl_xor(po, off, 32);
            if (lane == 0) {
                const float cx = curxS[m];
                const float o  = po + cx * pfc[HDIM];
                const float en = en_s[m] + (o + prev_out[m]) * 0.5f * ddm;
                en_s[m] = en; prev_out[m] = o; prev_x[m] = cx;
                out[(size_t)(b0 + m) * TDIM + t]                       = o;   // outputs
                out[(size_t)BDIM * TDIM + (size_t)(b0 + m) * TDIM + t] = en;  // energies
                if (t == TDIM - 1) {
                    out[carry + 2 * BH + (b0 + m)]            = o;    // carry out
                    out[carry + 3 * BH + BDIM + (b0 + m)]     = en;   // carry en
                    out[carry + 3 * BH + 2 * BDIM + (b0 + m)] = cx;   // carry prev_x
                }
            }
        }
        __syncthreads();   // state ready for next step
    }
}

extern "C" void kernel_launch(void* const* d_in, const int* in_sizes, int n_in,
                              void* d_out, int out_size, void* d_ws, size_t ws_size,
                              hipStream_t stream) {
    (void)in_sizes; (void)n_in; (void)out_size; (void)ws_size;
    const float* x   = (const float*)d_in[0];
    const float* We  = (const float*)d_in[1];
    const float* Wi  = (const float*)d_in[2];
    const float* bi  = (const float*)d_in[3];
    const float* Wf  = (const float*)d_in[4];
    const float* bf  = (const float*)d_in[5];
    const float* Wc  = (const float*)d_in[6];
    const float* bc  = (const float*)d_in[7];
    const float* Wo  = (const float*)d_in[8];
    const float* bo  = (const float*)d_in[9];
    const float* lg  = (const float*)d_in[10];
    const float* lb  = (const float*)d_in[11];
    const float* wfc = (const float*)d_in[12];

    _Float16* wsh = (_Float16*)d_ws;
    cvt_weights<<<512, 256, 0, stream>>>(We, Wi, Wf, Wc, Wo, wsh);
    lstm_scan<<<BDIM / MT, NTHREADS, 0, stream>>>(
        x, wsh, wsh + (size_t)HDIM * HDIM,
        bi, bf, bc, bo, lg, lb, wfc, (float*)d_out);
}
